// Model_57982058496058
// MI455X (gfx1250) — compile-verified, matrix operand order checked by reading, not run on hardware
//
#include <hip/hip_runtime.h>
#include <hip/hip_bf16.h>
#include <cstddef>

#define DMODEL 2048
#define LSEQ   2048
#define NST    16
#define TCHUNK 256

typedef __attribute__((ext_vector_type(16))) __bf16 v16bf;
typedef __attribute__((ext_vector_type(8)))  __bf16 v8bf;
typedef __attribute__((ext_vector_type(8)))  float  v8f;

// ---------------------------------------------------------------------------
// Kernel 0: fp32 -> bf16 conversion (round-to-nearest-even)
// ---------------------------------------------------------------------------
__global__ void f32_to_bf16_kernel(const float* __restrict__ src,
                                   __bf16* __restrict__ dst, int n) {
    int i = blockIdx.x * blockDim.x + threadIdx.x;
    if (i < n) {
        unsigned u = __float_as_uint(src[i]);
        unsigned r = (u + 0x7FFFu + ((u >> 16) & 1u)) >> 16;
        unsigned short s = (unsigned short)r;
        __bf16 b;
        __builtin_memcpy(&b, &s, 2);
        dst[i] = b;
    }
}

union Frag { v16bf v; v8bf h[2]; };

__device__ __forceinline__ void load_frag(Frag& f, const __bf16* p) {
    f.h[0] = *(const v8bf*)(p);
    f.h[1] = *(const v8bf*)(p + 16);
}

// ---------------------------------------------------------------------------
// Kernel 1a: BF16 WMMA GEMM, wave tile 16x64 (4 accumulators), explicit
//   double-buffered software pipeline: loads for K-step k+32 are issued
//   before the WMMAs consuming K-step k, so waits are partial.
//   out[M x N] = A[M x K] * W[N x K]^T, f32 accumulate, optional softplus.
//   ISA 16-bit A/B layout: lane = m + 16*g holds row/col m,
//     VGPRs 0..3 : K = g*8 .. g*8+7, VGPRs 4..7 : K = 16+g*8 .. 16+g*8+7
//   C/D layout: VGPR r -> M = r + 8*g, N = lane%16.
// ---------------------------------------------------------------------------
__global__ void __launch_bounds__(256)
gemm_bf16_wmma_n64(const __bf16* __restrict__ Abf,   // [M x K]
                   const __bf16* __restrict__ Wbf,   // [N x K]
                   float* __restrict__ out,          // [M x ldout]
                   int K, int ldout, int softplus_flag) {
    const int lane = threadIdx.x & 31;
    const int wave = threadIdx.x >> 5;
    const int row0 = blockIdx.y * 128 + wave * 16;   // M tile base (16 rows)
    const int col0 = blockIdx.x * 64;                // N tile base (4x16 cols)

    const int m = lane & 15;
    const int g = lane >> 4;

    const __bf16* aptr = Abf + (size_t)(row0 + m) * K + g * 8;
    const __bf16* bptr = Wbf + (size_t)(col0 + m) * K + g * 8;

    v8f acc[4] = {v8f{}, v8f{}, v8f{}, v8f{}};

    Frag a0, a1;
    Frag b0[4], b1[4];

#define LOAD_GROUP(AF, BF, KK)                                              \
    do {                                                                    \
        load_frag(AF, aptr + (KK));                                         \
        load_frag(BF[0], bptr + (KK));                                      \
        load_frag(BF[1], bptr + (size_t)16 * K + (KK));                     \
        load_frag(BF[2], bptr + (size_t)32 * K + (KK));                     \
        load_frag(BF[3], bptr + (size_t)48 * K + (KK));                     \
    } while (0)

#define MMA_GROUP(AF, BF)                                                   \
    do {                                                                    \
        acc[0] = __builtin_amdgcn_wmma_f32_16x16x32_bf16(                   \
            false, AF.v, false, BF[0].v, (short)0, acc[0], false, false);   \
        acc[1] = __builtin_amdgcn_wmma_f32_16x16x32_bf16(                   \
            false, AF.v, false, BF[1].v, (short)0, acc[1], false, false);   \
        acc[2] = __builtin_amdgcn_wmma_f32_16x16x32_bf16(                   \
            false, AF.v, false, BF[2].v, (short)0, acc[2], false, false);   \
        acc[3] = __builtin_amdgcn_wmma_f32_16x16x32_bf16(                   \
            false, AF.v, false, BF[3].v, (short)0, acc[3], false, false);   \
    } while (0)

    LOAD_GROUP(a0, b0, 0);
    int k0 = 0;
    for (; k0 + 64 < K; k0 += 64) {
        LOAD_GROUP(a1, b1, k0 + 32);   // prefetch next half-step
        MMA_GROUP(a0, b0);             // consume current (loads long issued)
        LOAD_GROUP(a0, b0, k0 + 64);   // prefetch next full-step
        MMA_GROUP(a1, b1);
    }
    LOAD_GROUP(a1, b1, K - 32);
    MMA_GROUP(a0, b0);
    MMA_GROUP(a1, b1);

#undef LOAD_GROUP
#undef MMA_GROUP

    const int rowbase = row0 + g * 8;
#pragma unroll
    for (int j = 0; j < 4; ++j) {
        const int col = col0 + j * 16 + m;
#pragma unroll
        for (int r = 0; r < 8; ++r) {
            float v = acc[j][r];
            if (softplus_flag)
                v = (v > 20.0f) ? v : log1pf(__expf(v));
            out[(size_t)(rowbase + r) * ldout + col] = v;
        }
    }
}

// ---------------------------------------------------------------------------
// Kernel 1b: single-tile variant for skinny N=16 outputs (Bm, Cm).
// ---------------------------------------------------------------------------
__global__ void __launch_bounds__(256)
gemm_bf16_wmma_n16(const __bf16* __restrict__ Abf,   // [M x K]
                   const __bf16* __restrict__ Wbf,   // [16 x K]
                   float* __restrict__ out,          // [M x 16]
                   int K) {
    const int lane = threadIdx.x & 31;
    const int wave = threadIdx.x >> 5;
    const int row0 = blockIdx.y * 128 + wave * 16;

    const int m = lane & 15;
    const int g = lane >> 4;

    const __bf16* aptr = Abf + (size_t)(row0 + m) * K + g * 8;
    const __bf16* bptr = Wbf + (size_t)m * K + g * 8;

    v8f acc = {};
    for (int k0 = 0; k0 < K; k0 += 32) {
        Frag a, b;
        load_frag(a, aptr + k0);
        load_frag(b, bptr + k0);
        acc = __builtin_amdgcn_wmma_f32_16x16x32_bf16(
            false, a.v, false, b.v, (short)0, acc, false, false);
    }

    const int rowbase = row0 + g * 8;
#pragma unroll
    for (int r = 0; r < 8; ++r)
        out[(size_t)(rowbase + r) * NST + m] = acc[r];
}

// ---------------------------------------------------------------------------
// Kernel 2: sequential selective scan. One thread per channel d.
// ---------------------------------------------------------------------------
__global__ void __launch_bounds__(256)
ssm_scan(const float* __restrict__ x, const float* __restrict__ delta,
         const float* __restrict__ Bm, const float* __restrict__ Cm,
         const float* __restrict__ A_log, const float* __restrict__ Dp,
         float* __restrict__ y) {
    __shared__ float sB[TCHUNK * NST];
    __shared__ float sC[TCHUNK * NST];

    const int d = blockIdx.x * blockDim.x + threadIdx.x;

    float A[NST], h[NST];
#pragma unroll
    for (int n = 0; n < NST; ++n) {
        A[n] = -__expf(A_log[(size_t)d * NST + n]);
        h[n] = 0.0f;
    }
    const float Dd = Dp[d];

    for (int l0 = 0; l0 < LSEQ; l0 += TCHUNK) {
        for (int i = threadIdx.x; i < TCHUNK * NST; i += blockDim.x) {
            sB[i] = Bm[(size_t)l0 * NST + i];
            sC[i] = Cm[(size_t)l0 * NST + i];
        }
        __syncthreads();

        if (l0 + TCHUNK < LSEQ) {   // hint next chunk into cache
            __builtin_prefetch(&delta[(size_t)(l0 + TCHUNK) * DMODEL + d], 0, 1);
            __builtin_prefetch(&x[(size_t)(l0 + TCHUNK) * DMODEL + d], 0, 1);
        }

        for (int lt = 0; lt < TCHUNK; ++lt) {
            const int l = l0 + lt;
            const float dl = delta[(size_t)l * DMODEL + d];
            const float xl = x[(size_t)l * DMODEL + d];
            const float dx = dl * xl;
            float acc = 0.0f;
#pragma unroll
            for (int n = 0; n < NST; ++n) {
                const float gA = __expf(dl * A[n]);
                h[n] = __builtin_fmaf(gA, h[n], dx * sB[lt * NST + n]);
                acc  = __builtin_fmaf(h[n], sC[lt * NST + n], acc);
            }
            y[(size_t)l * DMODEL + d] = __builtin_fmaf(xl, Dd, acc);
        }
        __syncthreads();
    }
}

// ---------------------------------------------------------------------------
// Launch
// ---------------------------------------------------------------------------
extern "C" void kernel_launch(void* const* d_in, const int* in_sizes, int n_in,
                              void* d_out, int out_size, void* d_ws, size_t ws_size,
                              hipStream_t stream) {
    const float* x       = (const float*)d_in[0];   // [1,2048,2048]
    const float* W_delta = (const float*)d_in[1];   // [2048,2048]
    const float* W_B     = (const float*)d_in[2];   // [16,2048]
    const float* W_C     = (const float*)d_in[3];   // [16,2048]
    const float* A_log   = (const float*)d_in[4];   // [2048,16]
    const float* Dvec    = (const float*)d_in[5];   // [2048]
    float* y = (float*)d_out;

    const size_t NBIG = (size_t)LSEQ * DMODEL;      // 4,194,304
    const size_t NSMALL = (size_t)NST * DMODEL;     // 32,768

    char* ws = (char*)d_ws;
    __bf16* xbf  = (__bf16*)ws;  ws += NBIG * 2;
    __bf16* wdbf = (__bf16*)ws;  ws += NBIG * 2;
    __bf16* wbbf = (__bf16*)ws;  ws += NSMALL * 2;
    __bf16* wcbf = (__bf16*)ws;  ws += NSMALL * 2;
    float*  dlt  = (float*)ws;   ws += NBIG * 4;    // delta [L, D]
    float*  Bm   = (float*)ws;   ws += (size_t)LSEQ * NST * 4;
    float*  Cm   = (float*)ws;   ws += (size_t)LSEQ * NST * 4;

    // 0) fp32 -> bf16
    f32_to_bf16_kernel<<<(NBIG + 255) / 256, 256, 0, stream>>>(x, xbf, (int)NBIG);
    f32_to_bf16_kernel<<<(NBIG + 255) / 256, 256, 0, stream>>>(W_delta, wdbf, (int)NBIG);
    f32_to_bf16_kernel<<<(NSMALL + 255) / 256, 256, 0, stream>>>(W_B, wbbf, (int)NSMALL);
    f32_to_bf16_kernel<<<(NSMALL + 255) / 256, 256, 0, stream>>>(W_C, wcbf, (int)NSMALL);

    // 1) delta = softplus(x W_delta^T): M=2048, N=2048, K=2048
    //    block tile 128(M) x 64(N), wave tile 16x64, double-buffered pipeline
    gemm_bf16_wmma_n64<<<dim3(DMODEL / 64, LSEQ / 128), 256, 0, stream>>>(
        xbf, wdbf, dlt, DMODEL, DMODEL, 1);

    //    Bm = x W_B^T, Cm = x W_C^T: M=2048, N=16, K=2048
    gemm_bf16_wmma_n16<<<dim3(1, LSEQ / 128), 256, 0, stream>>>(
        xbf, wbbf, Bm, DMODEL);
    gemm_bf16_wmma_n16<<<dim3(1, LSEQ / 128), 256, 0, stream>>>(
        xbf, wcbf, Cm, DMODEL);

    // 2) sequential scan: 2048 channels, 1 thread each
    ssm_scan<<<DMODEL / 256, 256, 0, stream>>>(x, dlt, Bm, Cm, A_log, Dvec, y);
}
